// MyBiLSTM_79645873537423
// MI455X (gfx1250) — matrix operand order repaired and checked by reading, not compiled, not run on hardware
//
#include <hip/hip_runtime.h>
#include <hip/hip_bf16.h>

typedef _Float16 half_t;
typedef __attribute__((ext_vector_type(8)))  _Float16 v8h;
typedef __attribute__((ext_vector_type(16))) _Float16 v16h;
typedef __attribute__((ext_vector_type(8)))  float    v8f;

#define SEQ  512
#define BSZ  64
#define HID  512
#define G4H  2048   // 4*HID
#define LROW 520    // LDS row stride in halves (512 + 8 pad -> conflict-free)

static __device__ __forceinline__ v16h cat16(v8h lo, v8h hi) {
  union { v16h v; v8h h[2]; } u;
  u.h[0] = lo; u.h[1] = hi;
  return u.v;
}

static __device__ __forceinline__ v8f wmma16(v16h a, v16h b, v8f c) {
  // D = A(16x32 f16) * B(32x16 f16) + C(16x16 f32)
  return __builtin_amdgcn_wmma_f32_16x16x32_f16(false, a, false, b, (short)0, c,
                                                false, false);
}

static __device__ __forceinline__ float sig_f(float x) {
  return 1.0f / (1.0f + __expf(-x));
}
static __device__ __forceinline__ float tanh_f(float x) {
  return 1.0f - 2.0f / (__expf(2.0f * x) + 1.0f);
}

// ---------------------------------------------------------------- converts ---
__global__ void k_cvt(const float* __restrict__ in, half_t* __restrict__ out, long n) {
  long i = (long)blockIdx.x * blockDim.x + threadIdx.x;
  long stride = (long)gridDim.x * blockDim.x;
  for (; i < n; i += stride) out[i] = (half_t)in[i];
}

// fuse biases (bih+bhh) per dir/layer and zero the grid-barrier words
__global__ void k_bias(const float* __restrict__ bihf, const float* __restrict__ bhhf,
                       const float* __restrict__ bihb, const float* __restrict__ bhhb,
                       float* __restrict__ bc, unsigned* __restrict__ bars) {
  int i = blockIdx.x * blockDim.x + threadIdx.x;
  if (i < 4096)       bc[i] = bihf[i] + bhhf[i];                 // dir0: [l][4H]
  else if (i < 8192)  bc[i] = bihb[i - 4096] + bhhb[i - 4096];   // dir1
  if (blockIdx.x == 0 && threadIdx.x < 64) bars[threadIdx.x] = 0u;
}

// ------------------------------------------------- input projection (GEMM) ---
// Xp[dir] holds fragment-packed fp16 gate tiles:
//   index = (((t*512 + tileId)*32 + lane)*8 + r), tileId = (g*32+ht)*4+bt
__global__ __launch_bounds__(256) void k_xproj(
    const half_t* __restrict__ A, size_t aDirStride,     // [S*B][512] fp16
    const half_t* __restrict__ W, size_t wDirStride,     // Wih[layer] rows=2048, K=512
    half_t* __restrict__ Xp, size_t xpDirStride) {
  const int dir = blockIdx.z;
  const int wv = threadIdx.x >> 5;
  const int l  = threadIdx.x & 31;
  const int hi = l >> 4, ln = l & 15;
  const int w   = blockIdx.x * 8 + wv;     // 65536 wave tasks
  const int s   = w >> 7;                  // 128 tasks per timestep
  const int rem = w & 127;
  const int ht  = rem >> 2, bt = rem & 3;

  const half_t* Ad = A + (size_t)dir * aDirStride;
  const half_t* Wd = W + (size_t)dir * wDirStride;
  const half_t* arow = Ad + ((size_t)s * BSZ + bt * 16 + ln) * HID + hi * 16;
  const half_t* b0r = Wd + (size_t)(0 * HID + ht * 16 + ln) * HID + hi * 16;
  const half_t* b1r = Wd + (size_t)(1 * HID + ht * 16 + ln) * HID + hi * 16;
  const half_t* b2r = Wd + (size_t)(2 * HID + ht * 16 + ln) * HID + hi * 16;
  const half_t* b3r = Wd + (size_t)(3 * HID + ht * 16 + ln) * HID + hi * 16;

  v8f acc0 = (v8f){0,0,0,0,0,0,0,0};
  v8f acc1 = acc0, acc2 = acc0, acc3 = acc0;

  // explicit double-buffered pipeline: fragments for kb+32 are loaded (and
  // live) across the WMMA group of kb, so the RA cannot alias the registers
  // and each wait covers loads issued one full group earlier.
  v16h a  = cat16(*(const v8h*)(arow + 0), *(const v8h*)(arow + 8));
  v16h b0 = cat16(*(const v8h*)(b0r + 0),  *(const v8h*)(b0r + 8));
  v16h b1 = cat16(*(const v8h*)(b1r + 0),  *(const v8h*)(b1r + 8));
  v16h b2 = cat16(*(const v8h*)(b2r + 0),  *(const v8h*)(b2r + 8));
  v16h b3 = cat16(*(const v8h*)(b3r + 0),  *(const v8h*)(b3r + 8));

#pragma unroll
  for (int kb = 0; kb < HID - 32; kb += 32) {
    const int nx = kb + 32;
    v16h an  = cat16(*(const v8h*)(arow + nx), *(const v8h*)(arow + nx + 8));
    v16h b0n = cat16(*(const v8h*)(b0r + nx),  *(const v8h*)(b0r + nx + 8));
    v16h b1n = cat16(*(const v8h*)(b1r + nx),  *(const v8h*)(b1r + nx + 8));
    v16h b2n = cat16(*(const v8h*)(b2r + nx),  *(const v8h*)(b2r + nx + 8));
    v16h b3n = cat16(*(const v8h*)(b3r + nx),  *(const v8h*)(b3r + nx + 8));
    acc0 = wmma16(a, b0, acc0);
    acc1 = wmma16(a, b1, acc1);
    acc2 = wmma16(a, b2, acc2);
    acc3 = wmma16(a, b3, acc3);
    a = an; b0 = b0n; b1 = b1n; b2 = b2n; b3 = b3n;
  }
  acc0 = wmma16(a, b0, acc0);
  acc1 = wmma16(a, b1, acc1);
  acc2 = wmma16(a, b2, acc2);
  acc3 = wmma16(a, b3, acc3);

  half_t* xp = Xp + (size_t)dir * xpDirStride;
  v8f accs[4] = {acc0, acc1, acc2, acc3};
#pragma unroll
  for (int g = 0; g < 4; ++g) {
    v8h o;
#pragma unroll
    for (int r = 0; r < 8; ++r) o[r] = (half_t)accs[g][r];
    size_t idx = (((size_t)s * 512 + (size_t)((g * 32 + ht) * 4 + bt)) * 32 + l) * 8;
    *(v8h*)(xp + idx) = o;   // coalesced b128 store
  }
}

// --------------------------------------------------------- recurrent phase ---
__global__ __launch_bounds__(256) void k_recur(
    const half_t* __restrict__ Xp,   size_t xpDirStride,
    const half_t* __restrict__ Whh,  size_t whhDirStride,   // [layer] rows=2048,K=512
    const float*  __restrict__ bias, size_t biasDirStride,  // [layer] 2048 floats
    half_t* __restrict__ Hs,         size_t hsDirStride,    // [layer] S*B*H fp16
    float* __restrict__ out,                                 // null for layer0
    unsigned* __restrict__ bars) {
  const int dir = blockIdx.y;
  constexpr unsigned NWG = 16;
  extern __shared__ half_t lds[];     // 128 rows x LROW halves = 133,120 B

  const int tid = threadIdx.x;
  const int wv = tid >> 5, l = tid & 31, hi = l >> 4, ln = l & 15;
  const int ht0 = blockIdx.x * 2;
  const int bt = wv & 3, hl = wv >> 2;
  const int ht = ht0 + hl;
  const int j  = ht * 16 + ln;        // hidden column owned by this lane
  const int batchBase = bt * 16;

  // ---- stage this WG's Whh slice (4 gates x 32 hidden cols -> 128 rows) ----
  const half_t* Wd = Whh + (size_t)dir * whhDirStride;
  {
    int rl = tid >> 1;                 // 128 local rows
    int hoff = (tid & 1) * 256;        // half-row per thread
    int g = rl >> 5, hl2 = (rl >> 4) & 1, n = rl & 15;
    const half_t* src = Wd + (size_t)(g * HID + (ht0 + hl2) * 16 + n) * HID + hoff;
    half_t* dst = lds + (size_t)rl * LROW + hoff;
#pragma unroll
    for (int i = 0; i < 256; i += 8)
      *(v8h*)(dst + i) = *(const v8h*)(src + i);
  }
  __syncthreads();

  const float* bd = bias + (size_t)dir * biasDirStride;
  const float bi_ = bd[j], bf_ = bd[HID + j], bg_ = bd[2 * HID + j], bo_ = bd[3 * HID + j];

  const half_t* xp = Xp + (size_t)dir * xpDirStride;
  half_t* hs = Hs + (size_t)dir * hsDirStride;
  unsigned* cnt = bars + dir * 8;
  unsigned* gen = bars + dir * 8 + 1;

  const half_t* lb0 = lds + (size_t)((0 * 2 + hl) * 16 + ln) * LROW + hi * 16;
  const half_t* lb1 = lds + (size_t)((1 * 2 + hl) * 16 + ln) * LROW + hi * 16;
  const half_t* lb2 = lds + (size_t)((2 * 2 + hl) * 16 + ln) * LROW + hi * 16;
  const half_t* lb3 = lds + (size_t)((3 * 2 + hl) * 16 + ln) * LROW + hi * 16;

  float c[8];
#pragma unroll
  for (int r = 0; r < 8; ++r) c[r] = 0.0f;

  for (int k = 0; k < SEQ; ++k) {
    const int t  = dir ? (SEQ - 1 - k) : k;
    const int tp = dir ? (t + 1) : (t - 1);

    // issue the long-latency Xproj fragment loads FIRST; loads complete in
    // order, so they are done long before the elementwise phase consumes them
    v8h xg0, xg1, xg2, xg3;
    {
      size_t tb = (size_t)t * 512;
      xg0 = *(const v8h*)(xp + ((tb + (size_t)((0 * 32 + ht) * 4 + bt)) * 32 + l) * 8);
      xg1 = *(const v8h*)(xp + ((tb + (size_t)((1 * 32 + ht) * 4 + bt)) * 32 + l) * 8);
      xg2 = *(const v8h*)(xp + ((tb + (size_t)((2 * 32 + ht) * 4 + bt)) * 32 + l) * 8);
      xg3 = *(const v8h*)(xp + ((tb + (size_t)((3 * 32 + ht) * 4 + bt)) * 32 + l) * 8);
    }

    v8f acc0 = (v8f){0,0,0,0,0,0,0,0};
    v8f acc1 = acc0, acc2 = acc0, acc3 = acc0;

    if (k > 0) {  // h_{-1} = 0: skip GEMM on first step
      const half_t* arow = hs + ((size_t)tp * BSZ + batchBase + ln) * HID + hi * 16;
      // double-buffered pipeline (A from global/L1, B from LDS)
      v16h a  = cat16(*(const v8h*)(arow + 0), *(const v8h*)(arow + 8));
      v16h b0 = cat16(*(const v8h*)(lb0 + 0),  *(const v8h*)(lb0 + 8));
      v16h b1 = cat16(*(const v8h*)(lb1 + 0),  *(const v8h*)(lb1 + 8));
      v16h b2 = cat16(*(const v8h*)(lb2 + 0),  *(const v8h*)(lb2 + 8));
      v16h b3 = cat16(*(const v8h*)(lb3 + 0),  *(const v8h*)(lb3 + 8));
#pragma unroll
      for (int kb = 0; kb < HID - 32; kb += 32) {
        const int nx = kb + 32;
        v16h an  = cat16(*(const v8h*)(arow + nx), *(const v8h*)(arow + nx + 8));
        v16h b0n = cat16(*(const v8h*)(lb0 + nx),  *(const v8h*)(lb0 + nx + 8));
        v16h b1n = cat16(*(const v8h*)(lb1 + nx),  *(const v8h*)(lb1 + nx + 8));
        v16h b2n = cat16(*(const v8h*)(lb2 + nx),  *(const v8h*)(lb2 + nx + 8));
        v16h b3n = cat16(*(const v8h*)(lb3 + nx),  *(const v8h*)(lb3 + nx + 8));
        acc0 = wmma16(a, b0, acc0);
        acc1 = wmma16(a, b1, acc1);
        acc2 = wmma16(a, b2, acc2);
        acc3 = wmma16(a, b3, acc3);
        a = an; b0 = b0n; b1 = b1n; b2 = b2n; b3 = b3n;
      }
      acc0 = wmma16(a, b0, acc0);
      acc1 = wmma16(a, b1, acc1);
      acc2 = wmma16(a, b2, acc2);
      acc3 = wmma16(a, b3, acc3);
    }

#pragma unroll
    for (int r = 0; r < 8; ++r) {
      float gi = acc0[r] + (float)xg0[r] + bi_;
      float gf = acc1[r] + (float)xg1[r] + bf_;
      float gg = acc2[r] + (float)xg2[r] + bg_;
      float go = acc3[r] + (float)xg3[r] + bo_;
      float iv = sig_f(gi), fv = sig_f(gf), gv = tanh_f(gg), ov = sig_f(go);
      float cn = fv * c[r] + iv * gv;
      c[r] = cn;
      float hv = ov * tanh_f(cn);
      int batch = batchBase + hi * 8 + r;       // C/D fragment row mapping
      hs[((size_t)t * BSZ + batch) * HID + j] = (half_t)hv;
      if (out)
        out[((size_t)t * BSZ + batch) * (2 * HID) + dir * HID + j] = hv;
    }

    if (k + 1 < SEQ) {
      // sense-reversing grid barrier over this direction's 16 workgroups
      __threadfence();
      __syncthreads();
      if (tid == 0) {
        unsigned g0 = __hip_atomic_load(gen, __ATOMIC_ACQUIRE, __HIP_MEMORY_SCOPE_AGENT);
        unsigned prev = __hip_atomic_fetch_add(cnt, 1u, __ATOMIC_ACQ_REL,
                                               __HIP_MEMORY_SCOPE_AGENT);
        if (prev == NWG - 1) {
          __hip_atomic_store(cnt, 0u, __ATOMIC_RELAXED, __HIP_MEMORY_SCOPE_AGENT);
          __hip_atomic_fetch_add(gen, 1u, __ATOMIC_ACQ_REL, __HIP_MEMORY_SCOPE_AGENT);
        } else {
          while (__hip_atomic_load(gen, __ATOMIC_ACQUIRE, __HIP_MEMORY_SCOPE_AGENT) == g0)
            __builtin_amdgcn_s_sleep(2);
        }
      }
      __syncthreads();
      __threadfence();
    }
  }
}

// ------------------------------------------------------------------- launch ---
extern "C" void kernel_launch(void* const* d_in, const int* in_sizes, int n_in,
                              void* d_out, int out_size, void* d_ws, size_t ws_size,
                              hipStream_t stream) {
  (void)in_sizes; (void)n_in; (void)out_size; (void)ws_size;
  const float* x     = (const float*)d_in[0];
  const float* Wih_f = (const float*)d_in[1];
  const float* Whh_f = (const float*)d_in[2];
  const float* bih_f = (const float*)d_in[3];
  const float* bhh_f = (const float*)d_in[4];
  const float* Wih_b = (const float*)d_in[5];
  const float* Whh_b = (const float*)d_in[6];
  const float* bih_b = (const float*)d_in[7];
  const float* bhh_b = (const float*)d_in[8];

  char* ws = (char*)d_ws;
  size_t off = 0;
  auto take = [&](size_t bytes) -> char* {
    char* p = ws + off;
    off = (off + bytes + 255) & ~(size_t)255;
    return p;
  };

  const size_t nX  = (size_t)SEQ * BSZ * HID;   // 16.78M elems
  const size_t nW  = (size_t)2 * G4H * HID;     // per-dir weight elems (L=2 layers)
  const size_t szL = (size_t)G4H * HID;         // per-layer weight elems
  const size_t xpS = (size_t)SEQ * BSZ * G4H;   // per-dir Xproj elems

  half_t*   xh    = (half_t*)take(nX * 2);
  half_t*   WihH  = (half_t*)take(2 * nW * 2);
  half_t*   WhhH  = (half_t*)take(2 * nW * 2);
  float*    biasC = (float*) take(2 * 2 * G4H * 4);    // [dir][layer][2048]
  half_t*   Hseq  = (half_t*)take(4 * nX * 2);         // [dir][layer][S*B*H]
  half_t*   Xp    = (half_t*)take(2 * xpS * 2);        // [dir][...]
  unsigned* bars  = (unsigned*)take(256);

  const int thr = 256;
  const size_t ldsBytes = (size_t)128 * LROW * 2;      // 133,120 B
  (void)hipFuncSetAttribute((const void*)k_recur,
                            hipFuncAttributeMaxDynamicSharedMemorySize,
                            (int)ldsBytes);

  // fp32 -> fp16 conversions
  k_cvt<<<dim3(2048), thr, 0, stream>>>(x, xh, (long)nX);
  k_cvt<<<dim3(1024), thr, 0, stream>>>(Wih_f, WihH,      (long)nW);
  k_cvt<<<dim3(1024), thr, 0, stream>>>(Wih_b, WihH + nW, (long)nW);
  k_cvt<<<dim3(1024), thr, 0, stream>>>(Whh_f, WhhH,      (long)nW);
  k_cvt<<<dim3(1024), thr, 0, stream>>>(Whh_b, WhhH + nW, (long)nW);
  k_bias<<<dim3(32), thr, 0, stream>>>(bih_f, bhh_f, bih_b, bhh_b, biasC, bars);

  // ---- layer 0 ----
  k_xproj<<<dim3(8192, 1, 2), thr, 0, stream>>>(xh, (size_t)0,
                                                WihH, nW, Xp, xpS);
  k_recur<<<dim3(16, 2), thr, ldsBytes, stream>>>(Xp, xpS,
                                                  WhhH, nW,
                                                  biasC, (size_t)(2 * G4H),
                                                  Hseq, (size_t)(2 * nX),
                                                  nullptr, bars);
  // ---- layer 1 ----
  k_xproj<<<dim3(8192, 1, 2), thr, 0, stream>>>(Hseq, (size_t)(2 * nX),
                                                WihH + szL, nW, Xp, xpS);
  k_recur<<<dim3(16, 2), thr, ldsBytes, stream>>>(Xp, xpS,
                                                  WhhH + szL, nW,
                                                  biasC + G4H, (size_t)(2 * G4H),
                                                  Hseq + nX, (size_t)(2 * nX),
                                                  (float*)d_out, bars);
}
